// GeoHCAN_90958817395314
// MI455X (gfx1250) — matrix operand order repaired and checked by reading, not compiled, not run on hardware
//
#include <hip/hip_runtime.h>
#include <hip/hip_bf16.h>

typedef _Float16 half_t;
typedef _Float16 v16h __attribute__((ext_vector_type(16)));
typedef float v8f __attribute__((ext_vector_type(8)));
typedef float v2f __attribute__((ext_vector_type(2)));
typedef int v4i_t __attribute__((vector_size(16)));

#define AS1 __attribute__((address_space(1)))
#define AS3 __attribute__((address_space(3)))

#if __has_builtin(__builtin_amdgcn_global_load_async_to_lds_b128) && \
    __has_builtin(__builtin_amdgcn_s_wait_asynccnt)
#define USE_ASYNC_LDS 1
#endif

#define B_  16
#define T_  4096
#define DM  1024
#define DP  64
#define TOK (B_*T_)
#define MT  128
#define NBLK (TOK/MT)

// ---- workspace float offsets ----
#define WS_SUM   0
#define WS_SUMSQ (16*1024)
#define WS_MEAN  (32*1024)
#define WS_INV   (48*1024)
#define WS_ACC   (64*1024)          // +0 tempsum, +1 cossum, +2 softplus(scale_logits), +3 softplus(scale_evidence)
#define WS_KN    (64*1024+8)        // 256 floats
#define WS_HALF_F32OFF (66*1024)    // half region starts here (float offset)
// half region (offsets in halves): WpT [64][1024] @0 ; WoT [1024][64] @65536 ; W2T [64][64] @131072

// ---- output offsets (floats) ----
#define O_USTD  67108864ull
#define O_TEMP  67174400ull
#define O_ATTN  67174401ull
#define O_ORTHO 67436545ull
#define O_GEO   67436546ull
#define O_LAT   67436547ull

// ---- shared memory layout (bytes) for main kernel ----
#define SM_QN    0        // 32KB float: Q -> Qn (in place) -> later Obuf
#define SM_HR    32768    // 32KB float: H_geo -> H_recon
#define SM_R3    65536    // 16KB: W1t (f32) then Ah (f16)
#define SM_R4    81920    // 16KB: W2s (first 8KB) then Wo chunk ping-pong (2x8KB)
#define SM_GATE  98304    // 512B
#define SM_RED   98816    // 512B
#define SM_KN    99328    // 1KB
#define SM_CE    100352   // 1KB
#define SM_A0    101376   // 2x8KB GEMM1 A tile ping-pong
#define SM_B0    117760   // 2x4KB GEMM1 B tile ping-pong
#define SM_T1    101376   // 16KB T1h (reuses A/B tile space after GEMM1)
#define SMEM_TOTAL 125952

__device__ __forceinline__ float softplusf(float v){ return v > 20.f ? v : log1pf(expf(v)); }
__device__ __forceinline__ float sigmoidf(float v){ return 1.f/(1.f+expf(-v)); }
__device__ __forceinline__ v8f vzero8(){ v8f z = {0.f,0.f,0.f,0.f,0.f,0.f,0.f,0.f}; return z; }

__device__ __forceinline__ v16h ld_frag16(const half_t* p0, const half_t* p1){
  union { float4 f[2]; v16h h; } u;
  u.f[0] = *(const float4*)p0;
  u.f[1] = *(const float4*)p1;
  return u.h;
}
__device__ __forceinline__ v2f ld_frag2(const float* p){
  union { float2 f; v2f v; } u;
  u.f = *(const float2*)p;
  return u.v;
}

// 16B global->LDS copy: async DMA (ASYNCcnt) when available, else through VGPRs
__device__ __forceinline__ void cp16(const half_t* g, half_t* l){
#if defined(USE_ASYNC_LDS)
  __builtin_amdgcn_global_load_async_to_lds_b128((AS1 v4i_t*)g, (AS3 v4i_t*)l, 0, 0);
#else
  *(float4*)l = *(const float4*)g;
#endif
}
__device__ __forceinline__ void async_wait_all(){
#if defined(USE_ASYNC_LDS)
  __builtin_amdgcn_s_wait_asynccnt(0);
#endif
}

// ---------------- helper kernels ----------------
__global__ void geo_zero(float* ws){
  int i = blockIdx.x * 256 + threadIdx.x;      // 32768 threads
  ws[i] = 0.f;                                  // zeros SUM + SUMSQ
}

__global__ void geo_stats1(const float* __restrict__ x, float* __restrict__ ws){
  int b  = blockIdx.x >> 4;
  int t0 = (blockIdx.x & 15) * 256;
  int tid = threadIdx.x;
  const float4* xp = (const float4*)x;
  size_t base = ((size_t)b * T_ + t0) * (DM/4) + tid;
  float4 s = {0,0,0,0}, q = {0,0,0,0};
  for (int t = 0; t < 256; ++t){
    float4 v = xp[base + (size_t)t * (DM/4)];
    s.x += v.x; s.y += v.y; s.z += v.z; s.w += v.w;
    q.x += v.x*v.x; q.y += v.y*v.y; q.z += v.z*v.z; q.w += v.w*v.w;
  }
  int d = b*DM + tid*4;
  atomicAdd(ws + WS_SUM   + d + 0, s.x); atomicAdd(ws + WS_SUM   + d + 1, s.y);
  atomicAdd(ws + WS_SUM   + d + 2, s.z); atomicAdd(ws + WS_SUM   + d + 3, s.w);
  atomicAdd(ws + WS_SUMSQ + d + 0, q.x); atomicAdd(ws + WS_SUMSQ + d + 1, q.y);
  atomicAdd(ws + WS_SUMSQ + d + 2, q.z); atomicAdd(ws + WS_SUMSQ + d + 3, q.w);
}

__global__ void geo_stats2(float* ws){
  int i = blockIdx.x * 256 + threadIdx.x;      // 16384
  float s = ws[WS_SUM + i], q = ws[WS_SUMSQ + i];
  float mean = s / (float)T_;
  float var  = fmaxf((q - s * mean) / (float)(T_ - 1), 0.f);  // unbiased (ddof=1)
  ws[WS_MEAN + i] = mean;
  ws[WS_INV  + i] = 1.f / (sqrtf(var) + 1e-5f);
}

__global__ void geo_pack(const float* __restrict__ Wp, const float* __restrict__ Wo,
                         const float* __restrict__ W2, half_t* __restrict__ wsh){
  int i = blockIdx.x * 256 + threadIdx.x;      // 65536
  { int c = i & 63, k = i >> 6; wsh[c*1024 + k] = (half_t)Wp[(size_t)k*64 + c]; }
  { int n = i >> 6, k = i & 63; wsh[65536 + n*64 + k] = (half_t)Wo[(size_t)k*1024 + n]; }
  if (i < 4096){ int c = i & 63, k = i >> 6; wsh[131072 + c*64 + k] = (half_t)W2[k*64 + c]; }
}

__global__ void geo_small(const float* __restrict__ cen, const float* __restrict__ sL,
                          const float* __restrict__ sE, float* __restrict__ ws,
                          float* __restrict__ out){
  __shared__ float nrm[4];
  __shared__ float kn[256];
  __shared__ float red[16];
  int t = threadIdx.x;   // 64 threads
  if (t < 4){
    float s = 0.f;
    for (int d = 0; d < 64; ++d){ float v = cen[t*64 + d]; s += v*v; }
    nrm[t] = fmaxf(sqrtf(s), 1e-12f);
  }
  __syncthreads();
  for (int c = 0; c < 4; ++c) kn[c*64 + t] = cen[c*64 + t] / nrm[c];
  __syncthreads();
  for (int c = 0; c < 4; ++c) ws[WS_KN + c*64 + t] = kn[c*64 + t];
  if (t < 16){
    int i = t >> 2, j = t & 3;
    float g = 0.f;
    for (int d = 0; d < 64; ++d) g += kn[i*64 + d] * kn[j*64 + d];
    float e = g - ((i == j) ? 1.f : 0.f);
    red[t] = e * e;
  }
  __syncthreads();
  if (t == 0){
    float s = 0.f;
    for (int k = 0; k < 16; ++k) s += red[k];
    out[O_ORTHO] = s / 16.f;
    out[O_GEO]   = 0.f;
    ws[WS_ACC+0] = 0.f; ws[WS_ACC+1] = 0.f;
    ws[WS_ACC+2] = softplusf(sL[0]);
    ws[WS_ACC+3] = softplusf(sE[0]);
  }
}

__global__ void geo_final(const float* __restrict__ ws, float* __restrict__ out){
  out[O_TEMP] = ws[WS_ACC+0] * (1.f/65536.f);
  out[O_LAT]  = 1.f - ws[WS_ACC+1] * (1.f/65536.f);
}

// ---------------- fused main kernel ----------------
__global__ void __launch_bounds__(256)
geo_main(const float* __restrict__ x, const float* __restrict__ bp,
         const float* __restrict__ ln_g, const float* __restrict__ ln_b,
         const float* __restrict__ cen, const float* __restrict__ bo,
         const float* __restrict__ W1, const float* __restrict__ b1,
         const float* __restrict__ b2, const float* __restrict__ Tthp,
         const float* __restrict__ Tslp, float* __restrict__ out,
         float* ws, const half_t* __restrict__ wsh)
{
  extern __shared__ unsigned char smem[];
  float*  Qn      = (float*)(smem + SM_QN);
  float*  Obuf    = (float*)(smem + SM_QN);     // reuse after latent pass
  float*  Hrec    = (float*)(smem + SM_HR);
  float*  W1t     = (float*)(smem + SM_R3);
  half_t* Ah      = (half_t*)(smem + SM_R3);    // reuse after offset GEMM
  half_t* W2s     = (half_t*)(smem + SM_R4);
  float*  rowGate = (float*)(smem + SM_GATE);
  float*  redf    = (float*)(smem + SM_RED);
  float*  KNs     = (float*)(smem + SM_KN);
  float*  CEs     = (float*)(smem + SM_CE);
  half_t* T1h     = (half_t*)(smem + SM_T1);    // reuse after GEMM1

  const half_t* wpT = wsh;
  const half_t* woT = wsh + 65536;
  const half_t* w2T = wsh + 131072;

  const int tid  = threadIdx.x;
  const int lane = tid & 31;
  const int wid  = tid >> 5;
  const int m15  = lane & 15;
  const size_t rowBase = (size_t)blockIdx.x * MT;
  const int bIdx = (int)(rowBase / T_);
  const float spL = ws[WS_ACC+2];
  const float spE = ws[WS_ACC+3];
  const float Tth = Tthp[0], Tsl = Tslp[0];

  // stage K_n and centroids
  KNs[tid] = ws[WS_KN + tid];
  CEs[tid] = cen[tid];

  // ---- Phase 1: Q = x_norm @ Wp  (f16 WMMA, double-buffered K stream) ----
  v8f acc0 = vzero8(), acc1 = vzero8(), acc2 = vzero8(), acc3 = vzero8();
  {
    const int rA = tid >> 1, sA = tid & 1;
    const float* xrow = x + (rowBase + rA) * (size_t)DM + sA*16;
    const float* mrow = ws + WS_MEAN + (size_t)bIdx * DM + sA*16;
    const float* irow = ws + WS_INV  + (size_t)bIdx * DM + sA*16;
    const int cB = tid >> 2, sB = tid & 3;

    auto fillA = [&](int buf, int kc){
      half_t* Abuf = (half_t*)(smem + SM_A0 + buf*8192);
      float xv[16], mm[16], iv[16];
      #pragma unroll
      for (int q = 0; q < 4; ++q){
        *(float4*)(xv + 4*q) = *(const float4*)(xrow + kc + 4*q);
        *(float4*)(mm + 4*q) = *(const float4*)(mrow + kc + 4*q);
        *(float4*)(iv + 4*q) = *(const float4*)(irow + kc + 4*q);
      }
      union { half_t h[16]; float4 f[2]; } u;
      #pragma unroll
      for (int q = 0; q < 16; ++q) u.h[q] = (half_t)((xv[q] - mm[q]) * iv[q]);
      *(float4*)(Abuf + rA*32 + sA*16 + 0) = u.f[0];
      *(float4*)(Abuf + rA*32 + sA*16 + 8) = u.f[1];
    };
    auto fillB = [&](int buf, int kc){
      half_t* Bbuf = (half_t*)(smem + SM_B0 + buf*4096);
      cp16(wpT + cB*1024 + kc + sB*8, Bbuf + cB*32 + sB*8);   // async DMA path
    };

    fillA(0, 0);
    fillB(0, 0);
    const int NK = DM / 32;
    for (int it = 0; it < NK; ++it){
      const int p = it & 1;
      async_wait_all();
      __syncthreads();
      if (it + 1 < NK){ fillA(1-p, (it+1)*32); fillB(1-p, (it+1)*32); }
      const half_t* Abase = (const half_t*)(smem + SM_A0 + p*8192);
      const half_t* Bbase = (const half_t*)(smem + SM_B0 + p*4096);
      const half_t* Ab = Abase + (wid*16 + m15)*32;
      const int hs = (lane >> 4) * 16;
      v16h af = ld_frag16(Ab + hs, Ab + hs + 8);
      const int koff = (lane >> 4) * 16;
      const half_t* Bb0 = Bbase + (0*16 + m15)*32 + koff;
      const half_t* Bb1 = Bbase + (1*16 + m15)*32 + koff;
      const half_t* Bb2 = Bbase + (2*16 + m15)*32 + koff;
      const half_t* Bb3 = Bbase + (3*16 + m15)*32 + koff;
      acc0 = __builtin_amdgcn_wmma_f32_16x16x32_f16(false, af, false, ld_frag16(Bb0, Bb0+8), (short)0, acc0, false, false);
      acc1 = __builtin_amdgcn_wmma_f32_16x16x32_f16(false, af, false, ld_frag16(Bb1, Bb1+8), (short)0, acc1, false, false);
      acc2 = __builtin_amdgcn_wmma_f32_16x16x32_f16(false, af, false, ld_frag16(Bb2, Bb2+8), (short)0, acc2, false, false);
      acc3 = __builtin_amdgcn_wmma_f32_16x16x32_f16(false, af, false, ld_frag16(Bb3, Bb3+8), (short)0, acc3, false, false);
    }
  }
  __syncthreads();
  // spill Q (+bp) to LDS
  {
    v8f a[4] = {acc0, acc1, acc2, acc3};
    #pragma unroll
    for (int nt = 0; nt < 4; ++nt){
      #pragma unroll
      for (int j = 0; j < 8; ++j){
        int row = wid*16 + ((lane < 16) ? j : j + 8);
        int col = nt*16 + m15;
        Qn[row*64 + col] = a[nt][j] + bp[col];
      }
    }
  }
  __syncthreads();

  // ---- Phase 2: LayerNorm over 64 + l2norm (in place) ----
  float tTemp = 0.f, tCos = 0.f;
  {
    const int r = tid >> 1, h = tid & 1;
    float* qr = Qn + r*64 + h*32;
    float s = 0.f, sq = 0.f;
    #pragma unroll
    for (int i = 0; i < 32; ++i){ float v = qr[i]; s += v; sq += v*v; }
    s += __shfl_xor(s, 1); sq += __shfl_xor(sq, 1);
    float mu = s * (1.f/64.f);
    float var = sq * (1.f/64.f) - mu * mu;
    float rstd = rsqrtf(var + 1e-5f);
    float l2 = 0.f;
    #pragma unroll
    for (int i = 0; i < 32; ++i){
      int col = h*32 + i;
      float q = (qr[i] - mu) * rstd * ln_g[col] + ln_b[col];
      qr[i] = q; l2 += q*q;
    }
    l2 += __shfl_xor(l2, 1);
    float inv = 1.f / fmaxf(sqrtf(l2), 1e-12f);
    #pragma unroll
    for (int i = 0; i < 32; ++i) qr[i] *= inv;
  }
  __syncthreads();

  // ---- Phase 3: sims / gates / softmax / H_geo ----
  {
    const int r = tid >> 1, h = tid & 1;
    const float* qr = Qn + r*64 + h*32;
    float sim[4];
    #pragma unroll
    for (int c = 0; c < 4; ++c){
      float s = 0.f;
      #pragma unroll
      for (int i = 0; i < 32; ++i) s += qr[i] * KNs[c*64 + h*32 + i];
      s += __shfl_xor(s, 1);
      sim[c] = s;
    }
    float thr = 0.25f * (sim[0] + sim[1] + sim[2] + sim[3]);
    float sumev = 0.f, maxev = -1e30f;
    #pragma unroll
    for (int c = 0; c < 4; ++c){
      float e = softplusf((sim[c] - thr) * 5.0f) * spE;
      sumev += e; maxev = fmaxf(maxev, e);
    }
    float ustd  = 4.f / (sumev + 4.f);
    float uresp = 1.f / (1.f + log1pf(maxev * 4.f));
    float Temp  = 0.1f + 0.9f * sigmoidf(Tsl * (uresp - Tth));
    float at[4], lm = -1e30f, es = 0.f;
    #pragma unroll
    for (int c = 0; c < 4; ++c){ at[c] = sim[c] * spL / Temp; lm = fmaxf(lm, at[c]); }
    #pragma unroll
    for (int c = 0; c < 4; ++c){ at[c] = expf(at[c] - lm); es += at[c]; }
    float ies = 1.f / es;
    #pragma unroll
    for (int c = 0; c < 4; ++c) at[c] *= ies;
    size_t grow = rowBase + r;
    if (h == 0){
      out[O_USTD + grow] = ustd;
      #pragma unroll
      for (int c = 0; c < 4; ++c) out[O_ATTN + grow*4 + c] = at[c];
      rowGate[r] = (1.f - uresp) * (1.f - uresp);
      tTemp += Temp;
    }
    #pragma unroll
    for (int i = 0; i < 32; ++i){
      int col = h*32 + i;
      Hrec[r*64 + col] = at[0]*CEs[col] + at[1]*CEs[64+col] + at[2]*CEs[128+col] + at[3]*CEs[192+col];
    }
  }
  __syncthreads();

  // ---- Phase 4a: T1 = tanh(Qn @ W1 + b1)  (f32 WMMA 16x16x4) ----
  #pragma unroll
  for (int e = 0; e < 16; ++e){
    int lin = tid*16 + e;                 // W1t[col][k] = W1[k][col]
    int c = lin >> 6, k = lin & 63;
    W1t[lin] = W1[k*64 + c];
  }
  cp16(w2T + tid*16,     W2s + tid*16);
  cp16(w2T + tid*16 + 8, W2s + tid*16 + 8);
  async_wait_all();
  __syncthreads();
  {
    v8f oa0 = vzero8(), oa1 = vzero8(), oa2 = vzero8(), oa3 = vzero8();
    const int ks2 = (lane >> 4) * 2;
    for (int k = 0; k < 64; k += 4){
      v2f a = ld_frag2(Qn + (wid*16 + m15)*64 + k + ks2);
      oa0 = __builtin_amdgcn_wmma_f32_16x16x4_f32(false, a, false, ld_frag2(W1t + (0*16+m15)*64 + k + ks2), (short)0, oa0, false, false);
      oa1 = __builtin_amdgcn_wmma_f32_16x16x4_f32(false, a, false, ld_frag2(W1t + (1*16+m15)*64 + k + ks2), (short)0, oa1, false, false);
      oa2 = __builtin_amdgcn_wmma_f32_16x16x4_f32(false, a, false, ld_frag2(W1t + (2*16+m15)*64 + k + ks2), (short)0, oa2, false, false);
      oa3 = __builtin_amdgcn_wmma_f32_16x16x4_f32(false, a, false, ld_frag2(W1t + (3*16+m15)*64 + k + ks2), (short)0, oa3, false, false);
    }
    v8f oa[4] = {oa0, oa1, oa2, oa3};
    #pragma unroll
    for (int nt = 0; nt < 4; ++nt){
      #pragma unroll
      for (int j = 0; j < 8; ++j){
        int row = wid*16 + ((lane < 16) ? j : j + 8);
        int col = nt*16 + m15;
        T1h[row*64 + col] = (half_t)tanhf(oa[nt][j] + b1[col]);
      }
    }
  }
  __syncthreads();

  // ---- Phase 4b: H_recon = H_geo + T1 @ W2 + b2 (f16 WMMA); also stash Ah f16 ----
  {
    v8f ob0 = vzero8(), ob1 = vzero8(), ob2 = vzero8(), ob3 = vzero8();
    #pragma unroll
    for (int ks = 0; ks < 64; ks += 32){
      const half_t* Ab = T1h + (wid*16 + m15)*64 + ks;
      const int hs = (lane >> 4) * 16;
      v16h af = ld_frag16(Ab + hs, Ab + hs + 8);
      const int koff = (lane >> 4) * 16;
      const half_t* Bb0 = W2s + (0*16+m15)*64 + ks + koff;
      const half_t* Bb1 = W2s + (1*16+m15)*64 + ks + koff;
      const half_t* Bb2 = W2s + (2*16+m15)*64 + ks + koff;
      const half_t* Bb3 = W2s + (3*16+m15)*64 + ks + koff;
      ob0 = __builtin_amdgcn_wmma_f32_16x16x32_f16(false, af, false, ld_frag16(Bb0, Bb0+8), (short)0, ob0, false, false);
      ob1 = __builtin_amdgcn_wmma_f32_16x16x32_f16(false, af, false, ld_frag16(Bb1, Bb1+8), (short)0, ob1, false, false);
      ob2 = __builtin_amdgcn_wmma_f32_16x16x32_f16(false, af, false, ld_frag16(Bb2, Bb2+8), (short)0, ob2, false, false);
      ob3 = __builtin_amdgcn_wmma_f32_16x16x32_f16(false, af, false, ld_frag16(Bb3, Bb3+8), (short)0, ob3, false, false);
    }
    v8f ob[4] = {ob0, ob1, ob2, ob3};
    #pragma unroll
    for (int nt = 0; nt < 4; ++nt){
      #pragma unroll
      for (int j = 0; j < 8; ++j){
        int row = wid*16 + ((lane < 16) ? j : j + 8);
        int col = nt*16 + m15;
        float v = ob[nt][j] + b2[col] + Hrec[row*64 + col];
        Hrec[row*64 + col] = v;
        Ah[row*64 + col] = (half_t)v;
      }
    }
  }
  __syncthreads();

  // prefetch first Wo chunk into BoS buffer 0 (overlaps with latent pass)
  auto fillBo = [&](int buf, int ncb){
    half_t* dst = (half_t*)(smem + SM_R4 + buf*8192);
    const half_t* src = woT + ncb*4096 + tid*16;
    cp16(src,     dst + tid*16);
    cp16(src + 8, dst + tid*16 + 8);
  };
  fillBo(0, 0);

  // ---- Phase 4c: latent cosine accumulation ----
  {
    const int r = tid >> 1, h = tid & 1;
    float d = 0.f, n1 = 0.f, n2 = 0.f;
    #pragma unroll
    for (int i = 0; i < 32; ++i){
      float hv = Hrec[r*64 + h*32 + i];
      float qv = Qn  [r*64 + h*32 + i];
      d += hv*qv; n1 += hv*hv; n2 += qv*qv;
    }
    d += __shfl_xor(d, 1); n1 += __shfl_xor(n1, 1); n2 += __shfl_xor(n2, 1);
    if (h == 0)
      tCos += d / (fmaxf(sqrtf(n1), 1e-8f) * fmaxf(sqrtf(n2), 1e-8f));
  }

  // ---- Phase 5: out = gate * (H_recon @ Wo + bo) + x  (f16 WMMA, double-buffered) ----
  for (int ncb = 0; ncb < 16; ++ncb){
    const int p = ncb & 1;
    async_wait_all();
    __syncthreads();
    if (ncb + 1 < 16) fillBo(1-p, ncb + 1);
    const half_t* BoB = (const half_t*)(smem + SM_R4 + p*8192);
    v8f oc0 = vzero8(), oc1 = vzero8(), oc2 = vzero8(), oc3 = vzero8();
    #pragma unroll
    for (int ks = 0; ks < 64; ks += 32){
      const half_t* Ab = Ah + (wid*16 + m15)*64 + ks;
      const int hs = (lane >> 4) * 16;
      v16h af = ld_frag16(Ab + hs, Ab + hs + 8);
      const int koff = (lane >> 4) * 16;
      const half_t* Bb0 = BoB + (0*16+m15)*64 + ks + koff;
      const half_t* Bb1 = BoB + (1*16+m15)*64 + ks + koff;
      const half_t* Bb2 = BoB + (2*16+m15)*64 + ks + koff;
      const half_t* Bb3 = BoB + (3*16+m15)*64 + ks + koff;
      oc0 = __builtin_amdgcn_wmma_f32_16x16x32_f16(false, af, false, ld_frag16(Bb0, Bb0+8), (short)0, oc0, false, false);
      oc1 = __builtin_amdgcn_wmma_f32_16x16x32_f16(false, af, false, ld_frag16(Bb1, Bb1+8), (short)0, oc1, false, false);
      oc2 = __builtin_amdgcn_wmma_f32_16x16x32_f16(false, af, false, ld_frag16(Bb2, Bb2+8), (short)0, oc2, false, false);
      oc3 = __builtin_amdgcn_wmma_f32_16x16x32_f16(false, af, false, ld_frag16(Bb3, Bb3+8), (short)0, oc3, false, false);
    }
    {
      v8f oc[4] = {oc0, oc1, oc2, oc3};
      #pragma unroll
      for (int nt = 0; nt < 4; ++nt){
        #pragma unroll
        for (int j = 0; j < 8; ++j){
          int row = wid*16 + ((lane < 16) ? j : j + 8);
          int col = nt*16 + m15;
          Obuf[row*64 + col] = oc[nt][j];
        }
      }
    }
    __syncthreads();
    #pragma unroll
    for (int e = 0; e < 8; ++e){
      int idx = (e*256 + tid) * 4;
      int row = idx >> 6, col = idx & 63;
      float4 o  = *(float4*)(Obuf + idx);
      float4 bv = *(const float4*)(bo + ncb*64 + col);
      size_t goff = (rowBase + row) * (size_t)DM + ncb*64 + col;
      float4 xv = *(const float4*)(x + goff);
      float g = rowGate[row];
      float4 r4;
      r4.x = g*(o.x + bv.x) + xv.x;
      r4.y = g*(o.y + bv.y) + xv.y;
      r4.z = g*(o.z + bv.z) + xv.z;
      r4.w = g*(o.w + bv.w) + xv.w;
      *(float4*)(out + goff) = r4;
    }
    __syncthreads();
  }

  // ---- Phase 6: block-reduce Temp / cos, atomic to ws ----
  #pragma unroll
  for (int m = 16; m >= 1; m >>= 1){
    tTemp += __shfl_xor(tTemp, m);
    tCos  += __shfl_xor(tCos,  m);
  }
  if (lane == 0){ redf[wid] = tTemp; redf[8 + wid] = tCos; }
  __syncthreads();
  if (tid == 0){
    float a = 0.f, c = 0.f;
    for (int w = 0; w < 8; ++w){ a += redf[w]; c += redf[8 + w]; }
    atomicAdd(ws + WS_ACC + 0, a);
    atomicAdd(ws + WS_ACC + 1, c);
  }
}

// ---------------- launch ----------------
extern "C" void kernel_launch(void* const* d_in, const int* in_sizes, int n_in,
                              void* d_out, int out_size, void* d_ws, size_t ws_size,
                              hipStream_t stream) {
  (void)in_sizes; (void)n_in; (void)out_size; (void)ws_size;
  const float* x    = (const float*)d_in[0];
  const float* Wp   = (const float*)d_in[1];
  const float* bp   = (const float*)d_in[2];
  const float* ln_g = (const float*)d_in[3];
  const float* ln_b = (const float*)d_in[4];
  const float* cen  = (const float*)d_in[5];
  const float* Wo   = (const float*)d_in[6];
  const float* bo   = (const float*)d_in[7];
  const float* W1   = (const float*)d_in[8];
  const float* b1   = (const float*)d_in[9];
  const float* W2   = (const float*)d_in[10];
  const float* b2   = (const float*)d_in[11];
  const float* sL   = (const float*)d_in[12];
  const float* sE   = (const float*)d_in[13];
  const float* Tth  = (const float*)d_in[14];
  const float* Tsl  = (const float*)d_in[15];
  float* out = (float*)d_out;
  float* ws  = (float*)d_ws;
  half_t* wsh = (half_t*)(ws + WS_HALF_F32OFF);

  geo_zero  <<<128, 256, 0, stream>>>(ws);
  geo_stats1<<<256, 256, 0, stream>>>(x, ws);
  geo_stats2<<<64,  256, 0, stream>>>(ws);
  geo_pack  <<<256, 256, 0, stream>>>(Wp, Wo, W2, wsh);
  geo_small <<<1,   64,  0, stream>>>(cen, sL, sE, ws, out);
  geo_main  <<<NBLK, 256, SMEM_TOTAL, stream>>>(x, bp, ln_g, ln_b, cen, bo, W1, b1, b2,
                                                Tth, Tsl, out, ws, wsh);
  geo_final <<<1, 1, 0, stream>>>(ws, out);
}
